// Spiral_MambaBlock_86569360818673
// MI455X (gfx1250) — matrix operand order repaired and checked by reading, not compiled, not run on hardware
//
#include <hip/hip_runtime.h>
#include <cstddef>
#include <cstdint>

// Problem constants (from reference): B=2, S=2048, D=512, E=1024, R=32, N=16
#define S_LEN 2048
#define D_DIM 512
#define E_DIM 1024
#define B_DIM 2
#define ROWS  (B_DIM * S_LEN)            // 4096
#define BSD   ((size_t)ROWS * D_DIM)     // 2,097,152
#define BSE   ((size_t)ROWS * E_DIM)     // 4,194,304

typedef __bf16 bf16_t;
typedef bf16_t v16bf __attribute__((ext_vector_type(16)));
typedef float  v8f   __attribute__((ext_vector_type(8)));
typedef unsigned int v4u __attribute__((ext_vector_type(4)));

union ABFrag { v16bf v; v4u u[2]; };

__device__ __forceinline__ unsigned short f2bf(float f) {
  unsigned u = __float_as_uint(f);
  u += 0x7FFFu + ((u >> 16) & 1u);          // round-to-nearest-even
  return (unsigned short)(u >> 16);
}
__device__ __forceinline__ float bf2f(unsigned short h) {
  return __uint_as_float(((unsigned)h) << 16);
}
__device__ __forceinline__ float silu_f(float v) { return v / (1.f + __expf(-v)); }

// CDNA5 async DMA: global -> LDS, 16B per lane, tracked by ASYNCcnt.
__device__ __forceinline__ void async_ld_b128(unsigned lds_off, const void* gptr) {
  asm volatile("global_load_async_to_lds_b128 %0, %1, off"
               :: "v"(lds_off), "v"(gptr) : "memory");
}
__device__ __forceinline__ void wait_async0() {
  asm volatile("s_wait_asynccnt 0x0" ::: "memory");
}

// ---------------------------------------------------------------------------
// Block reduction: sums 4 floats across the block. sred must hold >= 4*numWaves.
// ---------------------------------------------------------------------------
__device__ __forceinline__ void blk_reduce4(float& a, float& b, float& c, float& d,
                                            float* sred) {
  for (int o = 16; o; o >>= 1) {
    a += __shfl_xor(a, o); b += __shfl_xor(b, o);
    c += __shfl_xor(c, o); d += __shfl_xor(d, o);
  }
  int lane = threadIdx.x & 31, wid = threadIdx.x >> 5, nw = (blockDim.x + 31) >> 5;
  if (lane == 0) {
    sred[wid * 4 + 0] = a; sred[wid * 4 + 1] = b;
    sred[wid * 4 + 2] = c; sred[wid * 4 + 3] = d;
  }
  __syncthreads();
  if (threadIdx.x == 0) {
    float ta = 0, tb = 0, tc = 0, td = 0;
    for (int i = 0; i < nw; i++) {
      ta += sred[i * 4 + 0]; tb += sred[i * 4 + 1];
      tc += sred[i * 4 + 2]; td += sred[i * 4 + 3];
    }
    sred[0] = ta; sred[1] = tb; sred[2] = tc; sred[3] = td;
  }
  __syncthreads();
  a = sred[0]; b = sred[1]; c = sred[2]; d = sred[3];
  __syncthreads();
}

// ---------------------------------------------------------------------------
// WMMA bf16 GEMM:  C[M,N] = A[M,K] * Bw[N,K]^T  (+bias, +resid, +act)
// Block tile 128x128, 4 waves (2x2), wave tile 64x64 = 4x4 WMMA accums
// (16 ds_load_b128 : 16 wmma per K-step per wave). Double-buffered LDS;
// FULL path streams tiles with async-to-LDS DMA (ASYNCcnt), one barrier per
// K-step; !FULL path keeps guarded sync loads for partial tiles.
// ---------------------------------------------------------------------------
#define GEMM_BM 128
#define GEMM_BN 128
#define GEMM_BK 32
#define LSTR    40   // LDS row stride in bf16 elems (80B) -> conflict-free b128

template <bool HAS_BIAS, bool HAS_RESID, int ACT, bool OUT_F32, bool OUT_BF16,
          bool FULL>
__global__ void __launch_bounds__(128)
gemm_bf16(const unsigned short* __restrict__ A, const unsigned short* __restrict__ Bw,
          const float* __restrict__ bias, const float* __restrict__ resid,
          float* __restrict__ Cf, unsigned short* __restrict__ Cb,
          int M, int N, int K) {
  __shared__ unsigned short lA[2][GEMM_BM * LSTR];
  __shared__ unsigned short lB[2][GEMM_BN * LSTR];

  const int tid  = threadIdx.x;
  const int lane = tid & 31, wid = tid >> 5;
  const int m0 = blockIdx.y * GEMM_BM, n0 = blockIdx.x * GEMM_BN;
  const int waveM = wid & 1, waveN = wid >> 1;
  const int m0w = waveM * 64, n0w = waveN * 64;
  const int half = lane >> 4, l15 = lane & 15;

  // Staging: 128 rows x 4 quads (8 bf16 each) per tile; 4 slots per thread.
  const int rT = tid >> 2, qT = tid & 3;

  v8f acc[4][4];
#pragma unroll
  for (int mi = 0; mi < 4; mi++)
#pragma unroll
    for (int ni = 0; ni < 4; ni++) acc[mi][ni] = (v8f){0, 0, 0, 0, 0, 0, 0, 0};

  auto stage = [&](int buf, int k0) {
    if (FULL) {
#pragma unroll
      for (int p = 0; p < 4; p++) {
        int r = rT + p * 32;
        async_ld_b128((unsigned)(uintptr_t)&lA[buf][r * LSTR + qT * 8],
                      A + (size_t)(m0 + r) * K + k0 + qT * 8);
      }
#pragma unroll
      for (int p = 0; p < 4; p++) {
        int r = rT + p * 32;
        async_ld_b128((unsigned)(uintptr_t)&lB[buf][r * LSTR + qT * 8],
                      Bw + (size_t)(n0 + r) * K + k0 + qT * 8);
      }
    } else {
#pragma unroll
      for (int p = 0; p < 4; p++) {
        int r = rT + p * 32;
        v4u val = {0, 0, 0, 0};
        if (m0 + r < M) val = *(const v4u*)(A + (size_t)(m0 + r) * K + k0 + qT * 8);
        *(v4u*)&lA[buf][r * LSTR + qT * 8] = val;
      }
#pragma unroll
      for (int p = 0; p < 4; p++) {
        int r = rT + p * 32;
        v4u val = {0, 0, 0, 0};
        if (n0 + r < N) val = *(const v4u*)(Bw + (size_t)(n0 + r) * K + k0 + qT * 8);
        *(v4u*)&lB[buf][r * LSTR + qT * 8] = val;
      }
    }
  };

  stage(0, 0);
  int buf = 0;
  for (int k0 = 0; k0 < K; k0 += GEMM_BK, buf ^= 1) {
    if (FULL) wait_async0();            // my stage-k DMAs have landed in LDS
    __syncthreads();                    // everyone's landed; prev buffer free
    if (k0 + GEMM_BK < K) stage(buf ^ 1, k0 + GEMM_BK);

    // A fragments: lanes 0-15 hold K 0..7 & 16..23; lanes 16-31 hold 8..15 & 24..31
    ABFrag af[4];
#pragma unroll
    for (int mi = 0; mi < 4; mi++) {
      const unsigned short* base = &lA[buf][(m0w + mi * 16 + l15) * LSTR];
      af[mi].u[0] = *(const v4u*)(base + half * 8);
      af[mi].u[1] = *(const v4u*)(base + 16 + half * 8);
    }
    // B fragments: lanes 0-15 hold K 0..7 & 16..23; lanes 16-31 hold 8..15 & 24..31 (N = l15)
#pragma unroll
    for (int ni = 0; ni < 4; ni++) {
      ABFrag bfr;
      const unsigned short* base = &lB[buf][(n0w + ni * 16 + l15) * LSTR + half * 8];
      bfr.u[0] = *(const v4u*)(base);
      bfr.u[1] = *(const v4u*)(base + 16);
#pragma unroll
      for (int mi = 0; mi < 4; mi++)
        acc[mi][ni] = __builtin_amdgcn_wmma_f32_16x16x32_bf16(
            false, af[mi].v, false, bfr.v, (short)0, acc[mi][ni], false, false);
    }
  }

  // Epilogue: C layout — lane l, VGPR i => (m = i + 8*(l>>4), n = l&15)
#pragma unroll
  for (int ni = 0; ni < 4; ni++) {
    int col = n0 + n0w + ni * 16 + l15;
    float bv = 0.f;
    if (HAS_BIAS && (FULL || col < N)) bv = bias[col];
#pragma unroll
    for (int mi = 0; mi < 4; mi++) {
#pragma unroll
      for (int i = 0; i < 8; i++) {
        int gm = m0 + m0w + mi * 16 + i + 8 * half;
        if (FULL || (gm < M && col < N)) {
          float v = acc[mi][ni][i] + bv;
          if (HAS_RESID) v += resid[(size_t)gm * N + col];
          if (ACT == 1) v = silu_f(v);
          if (OUT_F32)  Cf[(size_t)gm * N + col] = v;
          if (OUT_BF16) Cb[(size_t)gm * N + col] = f2bf(v);
        }
      }
    }
  }
}

// ---------------------------------------------------------------------------
// f32 -> bf16 weight conversion
// ---------------------------------------------------------------------------
__global__ void k_f2bf(const float* __restrict__ src, unsigned short* __restrict__ dst,
                       int n) {
  int i = blockIdx.x * 256 + threadIdx.x;
  if (i < n) dst[i] = f2bf(src[i]);
}

// Wdbc (6,E,64) f32 -> transposed (6,64,E) bf16 (GEMM wants N-major, K contig.)
__global__ void k_wdbc_t(const float* __restrict__ src, unsigned short* __restrict__ dst) {
  int i = blockIdx.x * 256 + threadIdx.x;           // < 6*64*1024
  int dir = i >> 16, n = (i >> 10) & 63, e = i & 1023;
  dst[i] = f2bf(src[((size_t)(dir * 1024 + e)) * 64 + n]);
}

// ---------------------------------------------------------------------------
// mod = silu(c) @ W_ada^T + b_ada    (B, 3D) — tiny GEMM, scalar
// ---------------------------------------------------------------------------
__global__ void k_mod(const float* __restrict__ c, const float* __restrict__ W_ada,
                      const float* __restrict__ b_ada, float* __restrict__ mod) {
  int idx = blockIdx.x * 256 + threadIdx.x;          // < B*3D = 3072
  if (idx >= B_DIM * 3 * D_DIM) return;
  int b = idx / (3 * D_DIM), d3 = idx % (3 * D_DIM);
  float acc = 0.f;
  for (int k = 0; k < 2 * D_DIM; k++) {
    float cv = c[b * 2 * D_DIM + k];
    acc += silu_f(cv) * W_ada[(size_t)d3 * 2 * D_DIM + k];
  }
  mod[idx] = acc + b_ada[d3];
}

// ---------------------------------------------------------------------------
// Per-row: LN1(x)*(1+scale)+shift -> skip; *w -> wskip; LN2 -> xn; LN3 -> wn
// ---------------------------------------------------------------------------
__global__ void __launch_bounds__(128)
k_pre(const float* __restrict__ x, const float* __restrict__ w,
      const float* __restrict__ mod,
      const float* __restrict__ g1, const float* __restrict__ be1,
      const float* __restrict__ g2, const float* __restrict__ be2,
      const float* __restrict__ g3, const float* __restrict__ be3,
      float* __restrict__ skip, float* __restrict__ wskip,
      unsigned short* __restrict__ xn, unsigned short* __restrict__ wn) {
  __shared__ float sred[32];
  int row = blockIdx.x, tid = threadIdx.x;
  int b = row >> 11;                                  // S = 2048
  size_t rb = (size_t)row * D_DIM;

  float xv[4]; float s = 0, ss = 0, z0 = 0, z1 = 0;
#pragma unroll
  for (int j = 0; j < 4; j++) {
    int d = tid + j * 128;
    xv[j] = x[rb + d];
    s += xv[j]; ss += xv[j] * xv[j];
  }
  blk_reduce4(s, ss, z0, z1, sred);
  float m1 = s * (1.f / D_DIM), v1 = ss * (1.f / D_DIM) - m1 * m1;
  float rs1 = rsqrtf(v1 + 1e-5f);

  float xs[4], wsv[4]; float s2 = 0, ss2 = 0, s3 = 0, ss3 = 0;
#pragma unroll
  for (int j = 0; j < 4; j++) {
    int d = tid + j * 128;
    float ln1 = (xv[j] - m1) * rs1 * g1[d] + be1[d];
    float sc = mod[b * 3 * D_DIM + D_DIM + d];
    float sh = mod[b * 3 * D_DIM + d];
    float a = ln1 * (1.f + sc) + sh;
    float bw = a * w[rb + d];
    xs[j] = a; wsv[j] = bw;
    skip[rb + d] = a; wskip[rb + d] = bw;
    s2 += a; ss2 += a * a; s3 += bw; ss3 += bw * bw;
  }
  blk_reduce4(s2, ss2, s3, ss3, sred);
  float m2 = s2 * (1.f / D_DIM), v2 = ss2 * (1.f / D_DIM) - m2 * m2;
  float m3 = s3 * (1.f / D_DIM), v3 = ss3 * (1.f / D_DIM) - m3 * m3;
  float rs2 = rsqrtf(v2 + 1e-5f), rs3 = rsqrtf(v3 + 1e-5f);
#pragma unroll
  for (int j = 0; j < 4; j++) {
    int d = tid + j * 128;
    xn[rb + d] = f2bf((xs[j] - m2) * rs2 * g2[d] + be2[d]);
    wn[rb + d] = f2bf((wsv[j] - m3) * rs3 * g3[d] + be3[d]);
  }
}

// ---------------------------------------------------------------------------
// SSM scan, barrier-free. Grid (E/128, B, 6). Lane owns channel e: 16 states +
// 32-entry Wdt column in registers. The per-step dbc row is uniform across the
// wave (row index forced scalar via readfirstlane), so its 64 values come in
// as uniform/broadcast loads — no LDS, no barriers; Δ = softplus(dbc·Wdt) is
// recomputed in-register. Step-t output goes to row perm[t] ("origina" undo).
// ---------------------------------------------------------------------------
__global__ void __launch_bounds__(128)
k_scan(const float* __restrict__ zc, const float* __restrict__ wzc,
       const float* __restrict__ dbc_all, const float* __restrict__ Wdt,
       const float* __restrict__ bdt, const float* __restrict__ Alog,
       const int* __restrict__ order, const int* __restrict__ order_rev,
       unsigned short* __restrict__ ybuf) {
  const int dir = blockIdx.z, b = blockIdx.y;
  const int e = blockIdx.x * 128 + threadIdx.x;
  const int mode = dir % 3;
  const float* z = (dir < 3) ? zc : wzc;
  const float* dbc = dbc_all + (size_t)dir * ROWS * 64;
  unsigned short* y = ybuf + (size_t)dir * BSE;

  float wdt[32];
#pragma unroll
  for (int k = 0; k < 32; k++) wdt[k] = Wdt[((size_t)(dir * 32 + k)) * E_DIM + e];
  float A[16];
#pragma unroll
  for (int n = 0; n < 16; n++) A[n] = -__expf(Alog[((size_t)(dir * E_DIM + e)) * 16 + n]);
  const float bd = bdt[dir * E_DIM + e];
  float h[16];
#pragma unroll
  for (int n = 0; n < 16; n++) h[n] = 0.f;

  for (int t = 0; t < S_LEN; t++) {
    int row = (mode == 0) ? t : ((mode == 1) ? order[t] : order_rev[t]);
    row = __builtin_amdgcn_readfirstlane(row);        // uniform row index
    size_t rbase = (size_t)b * S_LEN + row;
    const float* drow = dbc + rbase * 64;             // uniform address

    float din = bd;
#pragma unroll
    for (int k = 0; k < 32; k++) din += drow[k] * wdt[k];
    float delta = (din > 20.f) ? din : log1pf(__expf(din));   // softplus
    float xv = z[rbase * E_DIM + e];
    float dx = delta * xv;
    float yv = 0.f;
#pragma unroll
    for (int n = 0; n < 16; n++) {
      h[n] = __expf(delta * A[n]) * h[n] + dx * drow[32 + n];
      yv += h[n] * drow[48 + n];
    }
    y[rbase * E_DIM + e] = f2bf(yv);
  }
}

// ---------------------------------------------------------------------------
// t = silu(mx) * ( y0 + y1 + y2 + z * (Dp0+Dp1+Dp2)[e] )  -> bf16
// ---------------------------------------------------------------------------
__global__ void k_combine(const float* __restrict__ mx, const float* __restrict__ zc,
                          const unsigned short* __restrict__ y0,
                          const unsigned short* __restrict__ y1,
                          const unsigned short* __restrict__ y2,
                          const float* __restrict__ Dp3,
                          unsigned short* __restrict__ tout) {
  size_t i = (size_t)blockIdx.x * 256 + threadIdx.x;
  int e = (int)(i & (E_DIM - 1));
  float ds = Dp3[e] + Dp3[E_DIM + e] + Dp3[2 * E_DIM + e];
  float yv = bf2f(y0[i]) + bf2f(y1[i]) + bf2f(y2[i]) + zc[i] * ds;
  tout[i] = f2bf(silu_f(mx[i]) * yv);
}

// ---------------------------------------------------------------------------
// comb = concat(x_out, w_out); LN(ga,ba) -> bf16 (feeds W1a GEMM)
// ---------------------------------------------------------------------------
__global__ void __launch_bounds__(256)
k_concat_ln(const float* __restrict__ xo, const float* __restrict__ wo,
            const float* __restrict__ ga, const float* __restrict__ ba,
            unsigned short* __restrict__ combn) {
  __shared__ float sred[32];
  int row = blockIdx.x, tid = threadIdx.x;
  float v[4]; float s = 0, ss = 0, z0 = 0, z1 = 0;
#pragma unroll
  for (int j = 0; j < 4; j++) {
    int d = tid + j * 256;
    v[j] = (d < D_DIM) ? xo[(size_t)row * D_DIM + d]
                       : wo[(size_t)row * D_DIM + (d - D_DIM)];
    s += v[j]; ss += v[j] * v[j];
  }
  blk_reduce4(s, ss, z0, z1, sred);
  float m = s * (1.f / (2 * D_DIM)), var = ss * (1.f / (2 * D_DIM)) - m * m;
  float rs = rsqrtf(var + 1e-5f);
#pragma unroll
  for (int j = 0; j < 4; j++) {
    int d = tid + j * 256;
    combn[(size_t)row * 2 * D_DIM + d] = f2bf((v[j] - m) * rs * ga[d] + ba[d]);
  }
}

// ---------------------------------------------------------------------------
// a = sigmoid(h . W2a + b2a);  out = x + gate * (a*x_out + (1-a)*w_out)
// ---------------------------------------------------------------------------
__global__ void __launch_bounds__(128)
k_final(const float* __restrict__ h, const float* __restrict__ W2a,
        const float* __restrict__ b2a, const float* __restrict__ xo,
        const float* __restrict__ wo, const float* __restrict__ x,
        const float* __restrict__ mod, float* __restrict__ out) {
  __shared__ float sred[32];
  int row = blockIdx.x, tid = threadIdx.x;
  int b = row >> 11;
  size_t rb = (size_t)row * D_DIM;
  float s = 0, z0 = 0, z1 = 0, z2 = 0;
#pragma unroll
  for (int j = 0; j < 4; j++) {
    int d = tid + j * 128;
    s += h[rb + d] * W2a[d];
  }
  blk_reduce4(s, z0, z1, z2, sred);
  float a = 1.f / (1.f + __expf(-(s + b2a[0])));
#pragma unroll
  for (int j = 0; j < 4; j++) {
    int d = tid + j * 128;
    float mix = a * xo[rb + d] + (1.f - a) * wo[rb + d];
    out[rb + d] = x[rb + d] + mod[b * 3 * D_DIM + 2 * D_DIM + d] * mix;
  }
}

// ---------------------------------------------------------------------------
extern "C" void kernel_launch(void* const* d_in, const int* in_sizes, int n_in,
                              void* d_out, int out_size, void* d_ws, size_t ws_size,
                              hipStream_t stream) {
  (void)in_sizes; (void)n_in; (void)out_size; (void)ws_size;
  const float* x     = (const float*)d_in[0];
  const float* c     = (const float*)d_in[1];
  const float* w     = (const float*)d_in[2];
  const float* W_ada = (const float*)d_in[3];
  const float* b_ada = (const float*)d_in[4];
  const float* g1 = (const float*)d_in[5],  *be1 = (const float*)d_in[6];
  const float* g2 = (const float*)d_in[7],  *be2 = (const float*)d_in[8];
  const float* g3 = (const float*)d_in[9],  *be3 = (const float*)d_in[10];
  const float* Wz  = (const float*)d_in[11], *bz  = (const float*)d_in[12];
  const float* Wx  = (const float*)d_in[13], *bx  = (const float*)d_in[14];
  const float* Wzw = (const float*)d_in[15], *bzw = (const float*)d_in[16];
  const float* Wxw = (const float*)d_in[17], *bxw = (const float*)d_in[18];
  const float* Wc  = (const float*)d_in[19], *bc  = (const float*)d_in[20];
  const float* Wcw = (const float*)d_in[21], *bcw = (const float*)d_in[22];
  const float* Wf  = (const float*)d_in[23], *bfv = (const float*)d_in[24];
  const float* Wfw = (const float*)d_in[25], *bfw = (const float*)d_in[26];
  const float* ga  = (const float*)d_in[27], *ba  = (const float*)d_in[28];
  const float* W1a = (const float*)d_in[29], *b1a = (const float*)d_in[30];
  const float* W2a = (const float*)d_in[31], *b2a = (const float*)d_in[32];
  const float* Wdbc = (const float*)d_in[33];
  const float* Wdt  = (const float*)d_in[34];
  const float* bdt  = (const float*)d_in[35];
  const float* Alog = (const float*)d_in[36];
  const float* Dssm = (const float*)d_in[37];
  const int* order     = (const int*)d_in[38];
  const int* order_rev = (const int*)d_in[39];
  float* out = (float*)d_out;

  // ---- workspace layout --------------------------------------------------
  size_t off = 0;
  auto alloc = [&](size_t bytes) -> void* {
    void* p = (char*)d_ws + off;
    off += (bytes + 255) & ~(size_t)255;
    return p;
  };
  float* mod   = (float*)alloc(B_DIM * 3 * D_DIM * 4);
  float* skip  = (float*)alloc(BSD * 4);
  float* wskip = (float*)alloc(BSD * 4);
  unsigned short* xn = (unsigned short*)alloc(BSD * 2);
  unsigned short* wn = (unsigned short*)alloc(BSD * 2);
  unsigned short* mz1 = (unsigned short*)alloc(BSE * 2);
  unsigned short* wz1 = (unsigned short*)alloc(BSE * 2);
  float* mx  = (float*)alloc(BSE * 4);
  float* wx  = (float*)alloc(BSE * 4);
  float* zc  = (float*)alloc(BSE * 4);
  float* wzc = (float*)alloc(BSE * 4);
  unsigned short* zcb  = (unsigned short*)alloc(BSE * 2);
  unsigned short* wzcb = (unsigned short*)alloc(BSE * 2);
  float* dbc = (float*)alloc((size_t)6 * ROWS * 64 * 4);
  unsigned short* ybuf = (unsigned short*)alloc((size_t)6 * BSE * 2);
  unsigned short* tx = (unsigned short*)alloc(BSE * 2);
  unsigned short* tw = (unsigned short*)alloc(BSE * 2);
  float* xo = (float*)alloc(BSD * 4);
  float* wo = (float*)alloc(BSD * 4);
  unsigned short* combn = (unsigned short*)alloc((size_t)ROWS * 2 * D_DIM * 2);
  float* hbuf = (float*)alloc(BSD * 4);
  // bf16 weights
  const int nED = E_DIM * D_DIM, nEE = E_DIM * E_DIM, nD2D = D_DIM * 2 * D_DIM;
  unsigned short* Wz_b  = (unsigned short*)alloc((size_t)nED * 2);
  unsigned short* Wx_b  = (unsigned short*)alloc((size_t)nED * 2);
  unsigned short* Wzw_b = (unsigned short*)alloc((size_t)nED * 2);
  unsigned short* Wxw_b = (unsigned short*)alloc((size_t)nED * 2);
  unsigned short* Wc_b  = (unsigned short*)alloc((size_t)nEE * 2);
  unsigned short* Wcw_b = (unsigned short*)alloc((size_t)nEE * 2);
  unsigned short* Wf_b  = (unsigned short*)alloc((size_t)nED * 2);
  unsigned short* Wfw_b = (unsigned short*)alloc((size_t)nED * 2);
  unsigned short* W1a_b = (unsigned short*)alloc((size_t)nD2D * 2);
  unsigned short* Wdbc_t = (unsigned short*)alloc((size_t)6 * 64 * E_DIM * 2);

  // ---- weight conversion -------------------------------------------------
  auto cvt = [&](const float* s, unsigned short* d, int n) {
    k_f2bf<<<(n + 255) / 256, 256, 0, stream>>>(s, d, n);
  };
  cvt(Wz, Wz_b, nED);   cvt(Wx, Wx_b, nED);
  cvt(Wzw, Wzw_b, nED); cvt(Wxw, Wxw_b, nED);
  cvt(Wc, Wc_b, nEE);   cvt(Wcw, Wcw_b, nEE);
  cvt(Wf, Wf_b, nED);   cvt(Wfw, Wfw_b, nED);
  cvt(W1a, W1a_b, nD2D);
  k_wdbc_t<<<(6 * 64 * E_DIM) / 256, 256, 0, stream>>>(Wdbc, Wdbc_t);

  // ---- conditioning + pre LNs -------------------------------------------
  k_mod<<<(B_DIM * 3 * D_DIM + 255) / 256, 256, 0, stream>>>(c, W_ada, b_ada, mod);
  k_pre<<<ROWS, 128, 0, stream>>>(x, w, mod, g1, be1, g2, be2, g3, be3,
                                  skip, wskip, xn, wn);

  // ---- input projections (WMMA, async-to-LDS staging) ---------------------
  dim3 blk(128);
  dim3 gE(E_DIM / 128, ROWS / 128);   // N=1024
  dim3 gD(D_DIM / 128, ROWS / 128);   // N=512
  dim3 g64(1, ROWS / 128);            // N=64 (partial tile)
  gemm_bf16<true, false, 0, false, true, true><<<gE, blk, 0, stream>>>(
      xn, Wz_b, bz, nullptr, nullptr, mz1, ROWS, E_DIM, D_DIM);
  gemm_bf16<true, false, 0, true, false, true><<<gE, blk, 0, stream>>>(
      xn, Wx_b, bx, nullptr, mx, nullptr, ROWS, E_DIM, D_DIM);
  gemm_bf16<true, false, 0, false, true, true><<<gE, blk, 0, stream>>>(
      wn, Wzw_b, bzw, nullptr, nullptr, wz1, ROWS, E_DIM, D_DIM);
  gemm_bf16<true, false, 0, true, false, true><<<gE, blk, 0, stream>>>(
      wn, Wxw_b, bxw, nullptr, wx, nullptr, ROWS, E_DIM, D_DIM);
  gemm_bf16<true, false, 0, true, true, true><<<gE, blk, 0, stream>>>(
      mz1, Wc_b, bc, nullptr, zc, zcb, ROWS, E_DIM, E_DIM);
  gemm_bf16<true, false, 0, true, true, true><<<gE, blk, 0, stream>>>(
      wz1, Wcw_b, bcw, nullptr, wzc, wzcb, ROWS, E_DIM, E_DIM);

  // ---- dbc projections per direction (WMMA, N=64, guarded partial tile) ---
  for (int dir = 0; dir < 6; dir++) {
    const unsigned short* Ain = (dir < 3) ? zcb : wzcb;
    gemm_bf16<false, false, 0, true, false, false><<<g64, blk, 0, stream>>>(
        Ain, Wdbc_t + (size_t)dir * 64 * E_DIM, nullptr, nullptr,
        dbc + (size_t)dir * ROWS * 64, nullptr, ROWS, 64, E_DIM);
  }

  // ---- channel-parallel SSM scans (6 dirs, barrier-free) ------------------
  k_scan<<<dim3(E_DIM / 128, B_DIM, 6), 128, 0, stream>>>(
      zc, wzc, dbc, Wdt, bdt, Alog, order, order_rev, ybuf);

  // ---- gated combine (silu(mx) * (Σ y + z * ΣD)) --------------------------
  k_combine<<<(unsigned)(BSE / 256), 256, 0, stream>>>(
      mx, zc, ybuf, ybuf + BSE, ybuf + 2 * BSE, Dssm, tx);
  k_combine<<<(unsigned)(BSE / 256), 256, 0, stream>>>(
      wx, wzc, ybuf + 3 * BSE, ybuf + 4 * BSE, ybuf + 5 * BSE,
      Dssm + 3 * E_DIM, tw);

  // ---- output projections + skip (WMMA) -----------------------------------
  gemm_bf16<true, true, 0, true, false, true><<<gD, blk, 0, stream>>>(
      tx, Wf_b, bfv, skip, xo, nullptr, ROWS, D_DIM, E_DIM);
  gemm_bf16<true, true, 0, true, false, true><<<gD, blk, 0, stream>>>(
      tw, Wfw_b, bfw, wskip, wo, nullptr, ROWS, D_DIM, E_DIM);

  // ---- adaptive mix -------------------------------------------------------
  k_concat_ln<<<ROWS, 256, 0, stream>>>(xo, wo, ga, ba, combn);
  gemm_bf16<true, false, 1, true, false, true><<<gD, blk, 0, stream>>>(
      combn, W1a_b, b1a, nullptr, hbuf, nullptr, ROWS, D_DIM, 2 * D_DIM);
  k_final<<<ROWS, 128, 0, stream>>>(hbuf, W2a, b2a, xo, wo, x, mod, out);
}
